// CauseMutiHeadAttention_17935783428651
// MI455X (gfx1250) — compile-verified, matrix operand order checked by reading, not run on hardware
//
#include <hip/hip_runtime.h>
#include <hip/hip_bf16.h>
#include <math.h>

// ---------------------------------------------------------------------------
// Causal multi-head attention for MI455X (gfx1250, wave32, WMMA).
// Attention uses transposed scores (S^T = K x Q^T) so softmax reductions are
// in-lane and P lands directly in WMMA A-layout (no LDS transpose).
// Two-pass flash: pass1 row max, pass2 exp + P*V accumulate.
// ---------------------------------------------------------------------------

typedef __attribute__((ext_vector_type(16))) _Float16 v16h;
typedef __attribute__((ext_vector_type(8)))  _Float16 v8h;
typedef __attribute__((ext_vector_type(4)))  _Float16 v4h;
typedef __attribute__((ext_vector_type(4)))  float    v4f;
typedef __attribute__((ext_vector_type(8)))  float    v8f;

#define D_MODEL 1024
#define NH      16
#define DK      64          // head dim
#define BB      4
#define SS      2048
#define NROW    (BB * SS)   // 8192 tokens

// ---------------------------------------------------------------------------
// WMMA helper: D = A(16x32 f16) x B(32x16 f16) + C (16x16 f32)
// ---------------------------------------------------------------------------
__device__ __forceinline__ v8f wmma_f16(v16h a, v16h b, v8f c) {
  return __builtin_amdgcn_wmma_f32_16x16x32_f16(
      /*neg_a=*/false, a, /*neg_b=*/false, b,
      /*c_mod=*/(short)0, c, /*reuse_a=*/false, /*reuse_b=*/false);
}

// ---------------------------------------------------------------------------
// Load a 16x32 f16 fragment from a row-major tile (A layout; B layout is the
// mirror with lane = column index, identical addressing).
// ---------------------------------------------------------------------------
__device__ __forceinline__ v16h load_frag16x32(const _Float16* __restrict__ base,
                                               int ld) {
  int lane = threadIdx.x & 31;
  int r    = lane & 15;
  int h    = lane >> 4;
  const _Float16* p = base + (size_t)r * ld + h * 8;
  v8h lo = *(const v8h*)(p);
  v8h hi = *(const v8h*)(p + 16);
  v16h f;
#pragma unroll
  for (int i = 0; i < 8; ++i) { f[i] = lo[i]; f[i + 8] = hi[i]; }
  return f;
}

// ---------------------------------------------------------------------------
// f32 -> f16 conversion, 4 elements/thread
// ---------------------------------------------------------------------------
__global__ void cvt_f32_to_f16_x4(const float* __restrict__ src,
                                  _Float16* __restrict__ dst, int n4) {
  int i = blockIdx.x * blockDim.x + threadIdx.x;
  if (i < n4) {
    v4f v = *(const v4f*)(src + (size_t)i * 4);
    v4h o;
#pragma unroll
    for (int j = 0; j < 4; ++j) o[j] = (_Float16)v[j];
    *(v4h*)(dst + (size_t)i * 4) = o;
  }
}

// ---------------------------------------------------------------------------
// GEMM: Y = A(f16, M x 1024 row-major) * W^T(f16, W 1024x1024 row-major) + b
// 64x32 output tile per wave (8 WMMAs / 6 fragment loads per K-step).
//   MODE 0: f16, head-major [B,H,S,64]   MODE 1: f16 transposed [B,H,64,S]
//   MODE 2: f32 row-major [M,1024]
// ---------------------------------------------------------------------------
template <int MODE>
__global__ void gemm_xwT(const _Float16* __restrict__ A,
                         const _Float16* __restrict__ W,
                         const float* __restrict__ bias,
                         void* __restrict__ out, int M) {
  const int NT = D_MODEL / 32;  // 32 column tiles
  int wave = (blockIdx.x * blockDim.x + threadIdx.x) >> 5;
  int row0 = (wave / NT) * 64;
  int col0 = (wave % NT) * 32;
  if (row0 >= M) return;  // wave-uniform

  v8f acc[4][2] = {};

  for (int k0 = 0; k0 < D_MODEL; k0 += 32) {
    v16h a0 = load_frag16x32(A + (size_t)(row0 +  0) * D_MODEL + k0, D_MODEL);
    v16h a1 = load_frag16x32(A + (size_t)(row0 + 16) * D_MODEL + k0, D_MODEL);
    v16h a2 = load_frag16x32(A + (size_t)(row0 + 32) * D_MODEL + k0, D_MODEL);
    v16h a3 = load_frag16x32(A + (size_t)(row0 + 48) * D_MODEL + k0, D_MODEL);
    v16h b0 = load_frag16x32(W + (size_t)col0 * D_MODEL + k0, D_MODEL);
    v16h b1 = load_frag16x32(W + (size_t)(col0 + 16) * D_MODEL + k0, D_MODEL);
    acc[0][0] = wmma_f16(a0, b0, acc[0][0]);
    acc[0][1] = wmma_f16(a0, b1, acc[0][1]);
    acc[1][0] = wmma_f16(a1, b0, acc[1][0]);
    acc[1][1] = wmma_f16(a1, b1, acc[1][1]);
    acc[2][0] = wmma_f16(a2, b0, acc[2][0]);
    acc[2][1] = wmma_f16(a2, b1, acc[2][1]);
    acc[3][0] = wmma_f16(a3, b0, acc[3][0]);
    acc[3][1] = wmma_f16(a3, b1, acc[3][1]);
  }

  int lane = threadIdx.x & 31;
  int n    = lane & 15;
  int h    = lane >> 4;

#pragma unroll
  for (int mi = 0; mi < 4; ++mi) {
#pragma unroll
    for (int ni = 0; ni < 2; ++ni) {
#pragma unroll
      for (int r = 0; r < 8; ++r) {
        int row = row0 + mi * 16 + r + 8 * h;   // token index
        int col = col0 + ni * 16 + n;           // feature index
        float v = acc[mi][ni][r] + bias[col];
        if (MODE == 2) {
          ((float*)out)[(size_t)row * D_MODEL + col] = v;
        } else {
          int bb = row >> 11;           // /SS
          int s  = row & (SS - 1);
          int hh = col >> 6;            // /DK
          int d  = col & (DK - 1);
          if (MODE == 0) {
            ((_Float16*)out)[((((size_t)bb * NH + hh) * SS) + s) * DK + d] =
                (_Float16)v;
          } else {  // MODE 1: V transposed [B,H,DK,S]
            ((_Float16*)out)[((((size_t)bb * NH + hh) * DK) + d) * SS + s] =
                (_Float16)v;
          }
        }
      }
    }
  }
}

// ---------------------------------------------------------------------------
// Transposed score tile: st[t][g] = K(keys kt+16g..+15, dims) x Q^T(queries
// q0+16t..+15).  D(m = key r+8h within group, n = query).
// ---------------------------------------------------------------------------
__device__ __forceinline__ void score_T(const _Float16* __restrict__ Kb,
                                        const v16h (&qb)[2][2], int kt,
                                        v8f (&st)[2][2]) {
  v16h ka[2][2];
#pragma unroll
  for (int g = 0; g < 2; ++g) {
    ka[g][0] = load_frag16x32(Kb + (size_t)(kt + 16 * g) * DK + 0,  DK);
    ka[g][1] = load_frag16x32(Kb + (size_t)(kt + 16 * g) * DK + 32, DK);
  }
#pragma unroll
  for (int t = 0; t < 2; ++t)
#pragma unroll
    for (int g = 0; g < 2; ++g) {
      st[t][g] = wmma_f16(ka[g][0], qb[t][0], st[t][g]);
      st[t][g] = wmma_f16(ka[g][1], qb[t][1], st[t][g]);
    }
}

// ---------------------------------------------------------------------------
// Flash attention (two-pass, transposed scores).  One wave per
// (b, h, 32-query tile).  Q,K: [B,H,S,64] f16.  Vt: [B,H,64,S] f16.
// Out: [B,S,H*64] f16 row-major.
// ---------------------------------------------------------------------------
__global__ __launch_bounds__(32)
void attn_causal(const _Float16* __restrict__ Q,
                 const _Float16* __restrict__ K,
                 const _Float16* __restrict__ Vt,
                 _Float16* __restrict__ Out) {
  __shared__ float Llds[32];

  int bid = blockIdx.x;
  int qt  = bid & (SS / 32 - 1);  // 64 query tiles
  int bh  = bid >> 6;             // b*NH + h
  int q0  = qt * 32;
  int b   = bh >> 4;
  int hd  = bh & 15;

  const _Float16* Qb = Q  + (size_t)bh * SS * DK;
  const _Float16* Kb = K  + (size_t)bh * SS * DK;
  const _Float16* Vb = Vt + (size_t)bh * DK * SS;

  int lane = threadIdx.x & 31;
  int n    = lane & 15;          // query within 16-tile (cols of S^T)
  int h    = lane >> 4;          // key half

  const float scale = 0.125f;    // 1/sqrt(64)

  // Q^T B-fragments: [query tile][dim chunk], loaded once
  v16h qb[2][2];
#pragma unroll
  for (int t = 0; t < 2; ++t) {
    qb[t][0] = load_frag16x32(Qb + (size_t)(q0 + 16 * t) * DK + 0,  DK);
    qb[t][1] = load_frag16x32(Qb + (size_t)(q0 + 16 * t) * DK + 32, DK);
  }

  // ------------------- pass 1: causal row max (in-lane) -------------------
  float mloc[2] = {-INFINITY, -INFINITY};
  for (int kt = 0; kt < q0; kt += 32) {
    __builtin_prefetch(Kb + (size_t)(kt + 32 + lane) * DK, 0, 0);
    v8f st[2][2] = {};
    score_T(Kb, qb, kt, st);
#pragma unroll
    for (int t = 0; t < 2; ++t)
#pragma unroll
      for (int g = 0; g < 2; ++g)
#pragma unroll
        for (int r = 0; r < 8; ++r) mloc[t] = fmaxf(mloc[t], st[t][g][r]);
  }
  {  // diagonal tile with causal mask
    v8f st[2][2] = {};
    score_T(Kb, qb, q0, st);
#pragma unroll
    for (int t = 0; t < 2; ++t) {
      int qrow = 16 * t + n;          // query - q0
#pragma unroll
      for (int g = 0; g < 2; ++g)
#pragma unroll
        for (int r = 0; r < 8; ++r) {
          int key = 16 * g + r + 8 * h;  // key - q0
          if (key <= qrow) mloc[t] = fmaxf(mloc[t], st[t][g][r]);
        }
    }
  }
  // merge the two key-halves (single cross-lane op for the whole pass)
  float ms[2];
#pragma unroll
  for (int t = 0; t < 2; ++t)
    ms[t] = fmaxf(mloc[t], __shfl_xor(mloc[t], 16, 32)) * scale;

  // ------------- pass 2: exp + row sum (in-lane) + P x V WMMAs ------------
  v8f   o[2][4] = {};
  float lloc[2] = {0.f, 0.f};

  for (int kt = 0; kt <= q0; kt += 32) {
    __builtin_prefetch(Kb + (size_t)(kt + 32 + lane) * DK, 0, 0);
    v8f st[2][2] = {};
    score_T(Kb, qb, kt, st);

    bool diag = (kt == q0);
    v16h pa[2];
#pragma unroll
    for (int t = 0; t < 2; ++t) {
      int qrow = 16 * t + n;
#pragma unroll
      for (int g = 0; g < 2; ++g)
#pragma unroll
        for (int r = 0; r < 8; ++r) {
          float s = st[t][g][r] * scale - ms[t];
          if (diag && (16 * g + r + 8 * h > qrow)) s = -INFINITY;
          float e = __expf(s);
          lloc[t] += e;
          // S^T D-layout == P A-layout: f[r] = key 8h+r (g=0),
          //                             f[8+r] = key 16+8h+r (g=1)
          pa[t][g * 8 + r] = (_Float16)e;
        }
    }
#pragma unroll
    for (int d = 0; d < 4; ++d) {
      v16h vb = load_frag16x32(Vb + (size_t)(d * 16) * SS + kt, SS);
      o[0][d] = wmma_f16(pa[0], vb, o[0][d]);
      o[1][d] = wmma_f16(pa[1], vb, o[1][d]);
    }
  }

  // combine halves of l and redistribute per O-row via tiny LDS stage
#pragma unroll
  for (int t = 0; t < 2; ++t) {
    float l = lloc[t] + __shfl_xor(lloc[t], 16, 32);
    Llds[16 * t + n] = l;   // both halves write identical values
  }
  __syncthreads();

  // ---- epilogue: normalize and store [B,S,H*64] f16 row-major ----
#pragma unroll
  for (int t = 0; t < 2; ++t)
#pragma unroll
    for (int r = 0; r < 8; ++r) {
      float inv = 1.0f / Llds[16 * t + r + 8 * h];
      int   s   = q0 + 16 * t + r + 8 * h;
      size_t base = ((size_t)b * SS + s) * D_MODEL + hd * DK;
#pragma unroll
      for (int d = 0; d < 4; ++d)
        Out[base + d * 16 + n] = (_Float16)(o[t][d][r] * inv);
    }
}

// ---------------------------------------------------------------------------
// Host-side launcher
// ---------------------------------------------------------------------------
extern "C" void kernel_launch(void* const* d_in, const int* in_sizes, int n_in,
                              void* d_out, int out_size, void* d_ws,
                              size_t ws_size, hipStream_t stream) {
  const float* x  = (const float*)d_in[0];
  const float* Wq = (const float*)d_in[1];
  const float* bq = (const float*)d_in[2];
  const float* Wk = (const float*)d_in[3];
  const float* bk = (const float*)d_in[4];
  const float* Wv = (const float*)d_in[5];
  const float* bv = (const float*)d_in[6];
  const float* Wo = (const float*)d_in[7];
  const float* bo = (const float*)d_in[8];

  char* ws = (char*)d_ws;
  const size_t SZ_X16 = (size_t)NROW * D_MODEL * sizeof(_Float16);     // 16 MB
  const size_t SZ_W16 = (size_t)D_MODEL * D_MODEL * sizeof(_Float16);  // 2 MB
  const size_t SZ_HB  = (size_t)BB * NH * SS * DK * sizeof(_Float16);  // 16 MB

  _Float16* x16  = (_Float16*)(ws);
  _Float16* wq16 = (_Float16*)(ws + SZ_X16);
  _Float16* wk16 = (_Float16*)(ws + SZ_X16 + SZ_W16);
  _Float16* wv16 = (_Float16*)(ws + SZ_X16 + 2 * SZ_W16);
  _Float16* wo16 = (_Float16*)(ws + SZ_X16 + 3 * SZ_W16);
  _Float16* Qb   = (_Float16*)(ws + SZ_X16 + 4 * SZ_W16);
  _Float16* Kb   = (_Float16*)(ws + SZ_X16 + 4 * SZ_W16 + SZ_HB);
  _Float16* Vtb  = (_Float16*)(ws + SZ_X16 + 4 * SZ_W16 + 2 * SZ_HB);
  _Float16* Ab   = (_Float16*)(ws + SZ_X16 + 4 * SZ_W16 + 3 * SZ_HB);

  // 1) f32 -> f16 conversions
  {
    int nx4 = NROW * D_MODEL / 4;
    cvt_f32_to_f16_x4<<<(nx4 + 255) / 256, 256, 0, stream>>>(x, x16, nx4);
    int nw4 = D_MODEL * D_MODEL / 4;
    cvt_f32_to_f16_x4<<<(nw4 + 255) / 256, 256, 0, stream>>>(Wq, wq16, nw4);
    cvt_f32_to_f16_x4<<<(nw4 + 255) / 256, 256, 0, stream>>>(Wk, wk16, nw4);
    cvt_f32_to_f16_x4<<<(nw4 + 255) / 256, 256, 0, stream>>>(Wv, wv16, nw4);
    cvt_f32_to_f16_x4<<<(nw4 + 255) / 256, 256, 0, stream>>>(Wo, wo16, nw4);
  }

  // 2) projections: waves = (NROW/64)*(D_MODEL/32) = 4096; 8 waves/block
  {
    int waves  = (NROW / 64) * (D_MODEL / 32);
    int blocks = waves / 8;  // 256 threads per block
    gemm_xwT<0><<<blocks, 256, 0, stream>>>(x16, wq16, bq, (void*)Qb,  NROW);
    gemm_xwT<0><<<blocks, 256, 0, stream>>>(x16, wk16, bk, (void*)Kb,  NROW);
    gemm_xwT<1><<<blocks, 256, 0, stream>>>(x16, wv16, bv, (void*)Vtb, NROW);
  }

  // 3) causal flash attention: one wave per (b, h, 32-query tile)
  attn_causal<<<BB * NH * (SS / 32), 32, 0, stream>>>(Qb, Kb, Vtb, Ab);

  // 4) output projection -> f32 d_out
  {
    int waves  = (NROW / 64) * (D_MODEL / 32);
    int blocks = waves / 8;
    gemm_xwT<2><<<blocks, 256, 0, stream>>>(Ab, wo16, bo, d_out, NROW);
  }
}
